// _ContextAttentionModule_79645873537062
// MI455X (gfx1250) — matrix operand order repaired and checked by reading, not compiled, not run on hardware
//
#include <hip/hip_runtime.h>
#include <hip/hip_bf16.h>
#include <stdint.h>

typedef __bf16 bf16_t;
typedef __attribute__((ext_vector_type(16))) __bf16 v16bf;
typedef __attribute__((ext_vector_type(8)))  float  v8f;

union BFrag { v16bf v; uint32_t u[8]; };
union PK    { __bf16 h[2]; uint32_t u; };

// ---------------- WMMA fragment helpers (CDNA5 16x16x32 bf16 layouts) ----------------
// A (16x32, bf16): lanes 0-15 row M=lane hold K = {0..7, 16..23}; lanes 16-31 row M=lane-16
// hold K = {8..15, 24..31}; 2 packed bf16 per VGPR.
__device__ __forceinline__ v16bf load_a_frag(const bf16_t* A, int lda, int kbase, unsigned lane) {
  const unsigned row = lane & 15u, half = lane >> 4;
  const bf16_t* p = A + (size_t)row * lda + kbase + half * 8;
  BFrag f;
#pragma unroll
  for (int j = 0; j < 4; ++j) f.u[j]     = *(const uint32_t*)(p + 2 * j);
#pragma unroll
  for (int j = 0; j < 4; ++j) f.u[4 + j] = *(const uint32_t*)(p + 16 + 2 * j);
  return f.v;
}

// B (32x16, bf16) sourced from memory stored as [col][K] (K contiguous):
// lanes 0-15 col N=lane hold K = {0..7, 16..23}; lanes 16-31 col N=lane-16 hold K = {8..15, 24..31}.
__device__ __forceinline__ v16bf load_b_frag(const bf16_t* BT, int ldb, int kbase, unsigned lane) {
  const unsigned col = lane & 15u, half = lane >> 4;
  const bf16_t* p = BT + (size_t)col * ldb + kbase + half * 8;
  BFrag f;
#pragma unroll
  for (int j = 0; j < 4; ++j) f.u[j]     = *(const uint32_t*)(p + 2 * j);
#pragma unroll
  for (int j = 0; j < 4; ++j) f.u[4 + j] = *(const uint32_t*)(p + 16 + 2 * j);
  return f.v;
}

__device__ __forceinline__ v8f wmma_bf16(v16bf a, v16bf b, v8f c) {
  return __builtin_amdgcn_wmma_f32_16x16x32_bf16(false, a, false, b, (short)0, c, false, false);
}

// ---------------- workspace layout (bytes) ----------------
#define OFF_X1T    ((size_t)0)              // bf16 [8][4096][256]   16.78 MB
#define OFF_X2T    ((size_t)16777216)       // bf16 [8][4096][256]
#define OFF_WG     ((size_t)33554432)       // bf16 [128][256]
#define OFF_WTH    ((size_t)33619968)
#define OFF_WPH    ((size_t)33685504)
#define OFF_WO     ((size_t)33751040)       // bf16 [256][128]
#define OFF_SUMS   ((size_t)33816576)       // f32 [512]  (sum, sumsq)
#define OFF_THETA  ((size_t)33818624)       // bf16 [8][4096][128]
#define OFF_PHIF   ((size_t)42207232)       // bf16 [8][4096][128]
#define OFF_GF     ((size_t)50595840)       // bf16 [8][4096][128]
#define OFF_PHIT   ((size_t)58984448)       // bf16 [8][1024][128]
#define OFF_GT     ((size_t)61081600)       // bf16 [8][128][1024]
#define OFF_YT     ((size_t)63178752)       // bf16 [8][4096][128]
#define OFF_Z      ((size_t)71567360)       // f32  [8][256][4096]  33.55 MB

// ---------------- kernel 0: weights -> bf16, zero BN accumulators ----------------
__global__ void prep_kernel(const float* wg, const float* wth, const float* wph, const float* wo,
                            bf16_t* wgb, bf16_t* wthb, bf16_t* wphb, bf16_t* wob, float* sums) {
  const int idx = blockIdx.x * blockDim.x + threadIdx.x;
  if (idx < 32768)            wgb[idx]           = (__bf16)wg[idx];
  else if (idx < 65536)       wthb[idx - 32768]  = (__bf16)wth[idx - 32768];
  else if (idx < 98304)       wphb[idx - 65536]  = (__bf16)wph[idx - 65536];
  else if (idx < 131072)      wob[idx - 98304]   = (__bf16)wo[idx - 98304];
  else if (idx < 131072+512)  sums[idx - 131072] = 0.0f;
}

// ---------------- kernel 1: transpose [b][c][n] f32 -> [b][n][c] bf16 (LDS tiled) ----------------
__global__ void __launch_bounds__(256) transpose_kernel(const float* __restrict__ x,
                                                        bf16_t* __restrict__ xt) {
  __shared__ float tile[64][65];
  const int t = threadIdx.x;
  const int bb = blockIdx.y;
  const int cblk = blockIdx.x >> 6;     // 0..3
  const int nblk = blockIdx.x & 63;     // 0..63
  const int r4 = t >> 6;                // 0..3
  const int cl = t & 63;
#pragma unroll 4
  for (int i = 0; i < 16; ++i) {
    const int r = i * 4 + r4;
    tile[r][cl] = x[(size_t)(bb * 256 + cblk * 64 + r) * 4096 + nblk * 64 + cl];
  }
  __syncthreads();
#pragma unroll 4
  for (int i = 0; i < 16; ++i) {
    const int nl = i * 4 + r4;
    xt[(size_t)(bb * 4096 + nblk * 64 + nl) * 256 + cblk * 64 + cl] = (__bf16)tile[cl][nl];
  }
}

// ---------------- kernel 2: 1x1 conv as GEMM  D[c,n] = W[c,:] @ Xt[n,:]^T + bias ----------------
// Output stored transposed [n][c] (contiguous packed b128 stores per lane).
__global__ void __launch_bounds__(256) conv1x1_kernel(const bf16_t* __restrict__ W,
                                                      const float* __restrict__ bias,
                                                      const bf16_t* __restrict__ Xt,
                                                      bf16_t* __restrict__ OutT) {
  const unsigned lane = threadIdx.x & 31u;
  const unsigned w    = threadIdx.x >> 5;     // wave -> c-tile (c_out = 128 = 8 tiles)
  const int bb = blockIdx.y;
  const int c0 = (int)w * 16;
  const bf16_t* Xb = Xt   + (size_t)bb * 4096 * 256;
  bf16_t*       Ob = OutT + (size_t)bb * 4096 * 128;
  const unsigned col = lane & 15u, half = lane >> 4;
#pragma unroll
  for (int nt = 0; nt < 4; ++nt) {
    const int n0 = blockIdx.x * 64 + nt * 16;
    v8f acc = {};
#pragma unroll
    for (int ks = 0; ks < 8; ++ks) {            // K = 256
      v16bf a = load_a_frag(W + (size_t)c0 * 256, 256, ks * 32, lane);
      v16bf b = load_b_frag(Xb + (size_t)n0 * 256, 256, ks * 32, lane);
      acc = wmma_bf16(a, b, acc);
    }
    uint32_t us[4];
#pragma unroll
    for (int j = 0; j < 4; ++j) {
      float lo = acc[2 * j]     + bias[c0 + half * 8 + 2 * j];
      float hi = acc[2 * j + 1] + bias[c0 + half * 8 + 2 * j + 1];
      PK pk; pk.h[0] = (__bf16)lo; pk.h[1] = (__bf16)hi;
      us[j] = pk.u;
    }
    uint4 q; q.x = us[0]; q.y = us[1]; q.z = us[2]; q.w = us[3];
    *(uint4*)(Ob + (size_t)(n0 + col) * 128 + c0 + half * 8) = q;   // transposed tile store
  }
}

// ---------------- kernel 3: 2x2 maxpool. mode 0: [n][c]->[m][c]; mode 1: [n][c]->[c][m] ----------
__global__ void pool_kernel(const bf16_t* __restrict__ in, bf16_t* __restrict__ out, int mode) {
  const int idx = blockIdx.x * blockDim.x + threadIdx.x;   // 8*1024*128 threads
  int c, m, bb;
  if (mode == 0) { c = idx & 127; m = (idx >> 7) & 1023; bb = idx >> 17; }
  else           { m = idx & 1023; c = (idx >> 10) & 127; bb = idx >> 17; }
  const int i = m >> 5, j = m & 31;
  const size_t base = (size_t)bb * 4096 * 128;
  const int n00 = (2 * i) * 64 + 2 * j;
  float a = (float)in[base + (size_t)(n00)      * 128 + c];
  float b = (float)in[base + (size_t)(n00 + 1)  * 128 + c];
  float e = (float)in[base + (size_t)(n00 + 64) * 128 + c];
  float d = (float)in[base + (size_t)(n00 + 65) * 128 + c];
  float mx = fmaxf(fmaxf(a, b), fmaxf(e, d));
  if (mode == 0) out[((size_t)bb * 1024 + m) * 128 + c] = (__bf16)mx;
  else           out[((size_t)bb * 128 + c) * 1024 + m] = (__bf16)mx;
}

// ---------------- kernel 4: fused attention  F = theta^T phi / M ;  Y' = F G^T ----------------
// 4 waves/WG, each wave owns a private 16x1024 F strip in LDS (never hits HBM).
#define FSTRIDE 1032
__global__ void __launch_bounds__(128) attn_kernel(const bf16_t* __restrict__ theta,
                                                   const bf16_t* __restrict__ phi,
                                                   const bf16_t* __restrict__ g,
                                                   bf16_t* __restrict__ yT) {
  __shared__ bf16_t Flds[4 * 16 * FSTRIDE];      // 132 KB (CDNA5 WGP: up to 320 KB)
  const unsigned lane = threadIdx.x & 31u;
  const unsigned w    = threadIdx.x >> 5;
  const int bb = blockIdx.y;
  const int n0 = blockIdx.x * 64 + (int)w * 16;
  const unsigned col = lane & 15u, half = lane >> 4;
  const bf16_t* thb = theta + (size_t)bb * 4096 * 128 + (size_t)n0 * 128;
  const bf16_t* phb = phi   + (size_t)bb * 1024 * 128;
  const bf16_t* gb  = g     + (size_t)bb * 128  * 1024;
  bf16_t*       yb  = yT    + (size_t)bb * 4096 * 128;
  bf16_t*       fw  = Flds + (size_t)w * 16 * FSTRIDE;

  // phase 1: 16x1024 F strip, scaled by 1/M, quantized to bf16 in LDS [n_local][m]
  for (int mt = 0; mt < 64; ++mt) {
    const int m0 = mt * 16;
    v8f acc = {};
#pragma unroll
    for (int ks = 0; ks < 4; ++ks) {             // K = 128 channels
      v16bf a = load_a_frag(thb, 128, ks * 32, lane);
      v16bf b = load_b_frag(phb + (size_t)m0 * 128, 128, ks * 32, lane);
      acc = wmma_bf16(a, b, acc);
    }
#pragma unroll
    for (int v = 0; v < 8; ++v)
      fw[(half * 8 + v) * FSTRIDE + m0 + col] = (__bf16)(acc[v] * (1.0f / 1024.0f));
  }

  // phase 2: Y'[n, c] = F[n, :] @ G^T[:, c]   (A from LDS, B from g [c][m])
  for (int ct = 0; ct < 8; ++ct) {
    const int c0 = ct * 16;
    v8f acc = {};
#pragma unroll 4
    for (int ks = 0; ks < 32; ++ks) {            // K = 1024
      v16bf a = load_a_frag(fw, FSTRIDE, ks * 32, lane);
      v16bf b = load_b_frag(gb + (size_t)c0 * 1024, 1024, ks * 32, lane);
      acc = wmma_bf16(a, b, acc);
    }
#pragma unroll
    for (int v = 0; v < 8; ++v)                  // store y as [n][c]
      yb[(size_t)(n0 + (int)half * 8 + v) * 128 + c0 + col] = (__bf16)acc[v];
  }
}

// ---------------- kernel 5: out conv + bias + BN partial sums ----------------
__global__ void __launch_bounds__(256) outconv_kernel(const bf16_t* __restrict__ Wout,
                                                      const float* __restrict__ bout,
                                                      const bf16_t* __restrict__ yT,
                                                      float* __restrict__ z,
                                                      float* gsum, float* gsq) {
  const unsigned lane = threadIdx.x & 31u;
  const unsigned w    = threadIdx.x >> 5;        // wave -> 32-channel strip of c_out=256
  const int bb = blockIdx.y;
  const unsigned col = lane & 15u, half = lane >> 4;
  const bf16_t* yb = yT + (size_t)bb * 4096 * 128;
  float*        zb = z  + (size_t)bb * 256 * 4096;
  float lsum[16] = {}, lsq[16] = {};
#pragma unroll 2
  for (int nt = 0; nt < 8; ++nt) {
    const int n0 = blockIdx.x * 128 + nt * 16;
#pragma unroll
    for (int cc = 0; cc < 2; ++cc) {
      const int c0 = (int)w * 32 + cc * 16;
      v8f acc = {};
#pragma unroll
      for (int ks = 0; ks < 4; ++ks) {           // K = 128
        v16bf a = load_a_frag(Wout + (size_t)c0 * 128, 128, ks * 32, lane);
        v16bf b = load_b_frag(yb + (size_t)n0 * 128, 128, ks * 32, lane);
        acc = wmma_bf16(a, b, acc);
      }
#pragma unroll
      for (int v = 0; v < 8; ++v) {
        const int ch = c0 + (int)half * 8 + v;
        float val = acc[v] + bout[ch];
        zb[(size_t)ch * 4096 + n0 + col] = val;  // z kept [c][n] (reference output order)
        lsum[cc * 8 + v] += val;
        lsq [cc * 8 + v] += val * val;
      }
    }
  }
#pragma unroll
  for (int cc = 0; cc < 2; ++cc)
#pragma unroll
    for (int v = 0; v < 8; ++v) {
      const int ch = (int)w * 32 + cc * 16 + (int)half * 8 + v;
      atomicAdd(&gsum[ch], lsum[cc * 8 + v]);
      atomicAdd(&gsq [ch], lsq [cc * 8 + v]);
    }
}

// ---------------- kernel 6: BN normalize + residual ----------------
__global__ void finalize_kernel(const float* __restrict__ z, const float* __restrict__ x1,
                                const float* gsum, const float* gsq,
                                const float* gamma, const float* beta, float* __restrict__ out) {
  const size_t idx = (size_t)blockIdx.x * blockDim.x + threadIdx.x;
  const int c = (int)((idx >> 12) & 255u);
  const float inv_cnt = 1.0f / 32768.0f;         // count = b * N = 8 * 4096
  const float mean = gsum[c] * inv_cnt;
  const float var  = gsq[c] * inv_cnt - mean * mean;
  const float s = rsqrtf(var + 1e-5f);
  out[idx] = (z[idx] - mean) * s * gamma[c] + beta[c] + x1[idx];
}

// ---------------- host launcher ----------------
extern "C" void kernel_launch(void* const* d_in, const int* in_sizes, int n_in,
                              void* d_out, int out_size, void* d_ws, size_t ws_size,
                              hipStream_t stream) {
  (void)in_sizes; (void)n_in; (void)out_size; (void)ws_size;
  const float* x1      = (const float*)d_in[0];
  const float* x2      = (const float*)d_in[1];
  const float* w_g     = (const float*)d_in[2];
  const float* b_g     = (const float*)d_in[3];
  const float* w_theta = (const float*)d_in[4];
  const float* b_theta = (const float*)d_in[5];
  const float* w_phi   = (const float*)d_in[6];
  const float* b_phi   = (const float*)d_in[7];
  const float* w_out   = (const float*)d_in[8];
  const float* b_out   = (const float*)d_in[9];
  const float* gamma   = (const float*)d_in[10];
  const float* beta    = (const float*)d_in[11];
  float* out = (float*)d_out;
  char* ws = (char*)d_ws;

  bf16_t* x1t   = (bf16_t*)(ws + OFF_X1T);
  bf16_t* x2t   = (bf16_t*)(ws + OFF_X2T);
  bf16_t* wgb   = (bf16_t*)(ws + OFF_WG);
  bf16_t* wthb  = (bf16_t*)(ws + OFF_WTH);
  bf16_t* wphb  = (bf16_t*)(ws + OFF_WPH);
  bf16_t* wob   = (bf16_t*)(ws + OFF_WO);
  float*  gsum  = (float*)(ws + OFF_SUMS);
  float*  gsq   = gsum + 256;
  bf16_t* thetaT= (bf16_t*)(ws + OFF_THETA);
  bf16_t* phiF  = (bf16_t*)(ws + OFF_PHIF);
  bf16_t* gF    = (bf16_t*)(ws + OFF_GF);
  bf16_t* phiT  = (bf16_t*)(ws + OFF_PHIT);
  bf16_t* gT    = (bf16_t*)(ws + OFF_GT);
  bf16_t* yT    = (bf16_t*)(ws + OFF_YT);
  float*  z     = (float*)(ws + OFF_Z);

  prep_kernel<<<514, 256, 0, stream>>>(w_g, w_theta, w_phi, w_out, wgb, wthb, wphb, wob, gsum);
  transpose_kernel<<<dim3(256, 8), 256, 0, stream>>>(x1, x1t);
  transpose_kernel<<<dim3(256, 8), 256, 0, stream>>>(x2, x2t);
  conv1x1_kernel<<<dim3(64, 8), 256, 0, stream>>>(wthb, b_theta, x1t, thetaT);
  conv1x1_kernel<<<dim3(64, 8), 256, 0, stream>>>(wphb, b_phi,   x2t, phiF);
  conv1x1_kernel<<<dim3(64, 8), 256, 0, stream>>>(wgb,  b_g,     x2t, gF);
  pool_kernel<<<4096, 256, 0, stream>>>(phiF, phiT, 0);
  pool_kernel<<<4096, 256, 0, stream>>>(gF,   gT,   1);
  attn_kernel<<<dim3(64, 8), 128, 0, stream>>>(thetaT, phiT, gT, yT);
  outconv_kernel<<<dim3(32, 8), 256, 0, stream>>>(wob, b_out, yT, z, gsum, gsq);
  finalize_kernel<<<32768, 256, 0, stream>>>(z, x1, gsum, gsq, gamma, beta, out);
}
